// ChannelWiseLSTM_19061064860394
// MI455X (gfx1250) — compile-verified, matrix operand order checked by reading, not run on hardware
//
#include <hip/hip_runtime.h>
#include <hip/hip_bf16.h>

// ---------------------------------------------------------------------------
// ChannelWiseLSTM for MI455X (gfx1250, wave32, WMMA)
//
//   k_chan_lstm : 256 independent single-wave chains (32 channels x 8 row
//                 blocks). Each wave owns 16 batch rows: recurrent GEMM
//                 h[16x32] @ Whh^T[32x128] as 8x v_wmma_f32_16x16x32_f16
//                 (K=32 == H, one WMMA per 16x16 tile), Whh register-resident
//                 as f16 B-tiles, c-state in f32 registers, h bounced through
//                 a 1KB LDS tile to rebuild the WMMA A layout each step.
//                 Activations use native v_tanh (CDNA5 TRANS op) when the
//                 builtin is available; x loads software-pipelined one step
//                 ahead. Channel-backward direction = one zero-state cell on
//                 x[:,-1]. Output written as f16 in merge layout [B][64][C].
//   k_merge_lstm: 8 single-wave chains over 64 steps; per step two chained
//                 WMMAs (x-GEMM feeding h-GEMM accumulator), A_x pipelined.
//                 Backward cell is one more WMMA batch on step 63.
//   k_fc        : 128x25 sigmoid(dot64) epilogue.
// ---------------------------------------------------------------------------

typedef __attribute__((ext_vector_type(16))) _Float16 v16h;
typedef __attribute__((ext_vector_type(8)))  _Float16 v8h;
typedef __attribute__((ext_vector_type(8)))  float    v8f;

namespace {
constexpr int B    = 128;
constexpr int T    = 512;
constexpr int C    = 32;
constexpr int H    = 32;
constexpr int G4H  = 4 * H;      // 128 gate columns
constexpr int NCLS = 25;
constexpr int SEQM = 2 * H;      // 64: merge-LSTM sequence length
}

// Native CDNA5 v_tanh when available; exp fallback otherwise.
__device__ __forceinline__ float ftanh(float v) {
#if __has_builtin(__builtin_amdgcn_tanhf)
  return __builtin_amdgcn_tanhf(v);
#elif __has_builtin(__builtin_amdgcn_tanh_f32)
  return __builtin_amdgcn_tanh_f32(v);
#else
  return 2.0f / (1.0f + __expf(-2.0f * v)) - 1.0f;
#endif
}
__device__ __forceinline__ float sigm(float v) {
  return __builtin_fmaf(ftanh(0.5f * v), 0.5f, 0.5f);
}

__device__ __forceinline__ float lane_bcast(float v, int byteidx) {
  return __int_as_float(__builtin_amdgcn_ds_bpermute(byteidx, __float_as_int(v)));
}

// Load one 16x16 B-tile (K=32 halves per lane-column) from a [128 x 32] f32
// row-major weight matrix (row = gate column, inner = K). Layout mirrors the
// documented 16-bit A layout: lanes 0-15 -> K {0..7,16..23}, lanes 16-31 ->
// K {8..15,24..31}.
__device__ __forceinline__ v16h load_btile(const float* __restrict__ w,
                                           int n, int lc, int kb) {
  const float* wrow = w + (n * 16 + lc) * 32;
  v16h t;
#pragma unroll
  for (int q = 0; q < 8; ++q) {
    t[q]     = (_Float16)wrow[kb + q];
    t[8 + q] = (_Float16)wrow[kb + 16 + q];
  }
  return t;
}

// ===========================================================================
// Kernel 1: per-channel BiLSTM (forward scan + backward single cell)
// grid = (C, B/16), block = 32 (one wave, no barriers needed)
// ===========================================================================
__global__ __launch_bounds__(32) void k_chan_lstm(
    const float* __restrict__ x,        // [B,T,C]
    const float* __restrict__ WihF,     // [C,128,1]
    const float* __restrict__ WhhF,     // [C,128,32]
    const float* __restrict__ bihF,     // [C,128]
    const float* __restrict__ bhhF,     // [C,128]
    const float* __restrict__ WihB,     // [C,128,1]
    const float* __restrict__ bihB,     // [C,128]
    const float* __restrict__ bhhB,     // [C,128]
    _Float16* __restrict__ Xm)          // [B,64,C] f16 (merge input layout)
{
  const int c       = blockIdx.x;
  const int rowbase = blockIdx.y * 16;
  const int lane    = threadIdx.x & 31;
  const int lc      = lane & 15;
  const int kb      = (lane >> 4) * 8;   // 0 or 8

  __shared__ _Float16 hbuf[16 * 32];     // h tile, [row][k], f16

  // hoisted bpermute byte-indices: source lane r+kb
  int bperm[8];
#pragma unroll
  for (int r = 0; r < 8; ++r) bperm[r] = (r + kb) << 2;

  // --- register-resident recurrent weights: 8 B-tiles of Whh^T -------------
  v16h bt[8];
  {
    const float* whh = WhhF + (size_t)c * G4H * H;
#pragma unroll
    for (int n = 0; n < 8; ++n) bt[n] = load_btile(whh, n, lc, kb);
  }

  // --- per-lane gate constants (columns n*16 + lc) -------------------------
  float wih[8], bia[8], wihb[8], biab[8];
#pragma unroll
  for (int n = 0; n < 8; ++n) {
    const int col = c * G4H + n * 16 + lc;
    wih[n]  = WihF[col];
    bia[n]  = bihF[col] + bhhF[col];
    wihb[n] = WihB[col];
    biab[n] = bihB[col] + bhhB[col];
  }

  // --- state init ----------------------------------------------------------
  float cs0[8], cs1[8];
#pragma unroll
  for (int r = 0; r < 8; ++r) {
    cs0[r] = 0.0f; cs1[r] = 0.0f;
    const int M = r + kb;
    hbuf[M * 32 + lc]      = (_Float16)0.0f;
    hbuf[M * 32 + lc + 16] = (_Float16)0.0f;
  }
  __syncthreads();

  const size_t xrow = (size_t)(rowbase + lc) * (T * C) + c;

  // ===== forward scan over T=512 steps (x pipelined one step ahead) ========
  float xv = x[xrow];
  for (int t = 0; t < T; ++t) {
    const int tn = (t + 1 < T) ? (t + 1) : (T - 1);
    const float xnext = x[xrow + (size_t)tn * C];     // covers next iteration

    // A tile: h in documented 16-bit A layout (two ds_load_b128)
    const _Float16* hrow = hbuf + lc * 32 + kb;
    v8h a0 = *(const v8h*)(hrow);
    v8h a1 = *(const v8h*)(hrow + 16);
    v16h a;
#pragma unroll
    for (int q = 0; q < 8; ++q) { a[q] = a0[q]; a[8 + q] = a1[q]; }

    // gates[16x128] = h @ Whh^T : one WMMA per N-tile (K=32 in one shot)
    v8f acc[8];
#pragma unroll
    for (int n = 0; n < 8; ++n) {
      v8f z = {};
      acc[n] = __builtin_amdgcn_wmma_f32_16x16x32_f16(
          false, a, false, bt[n], (short)0, z, false, false);
    }
    __syncthreads();

    // gate math + state update; rewrite h tile
#pragma unroll
    for (int r = 0; r < 8; ++r) {
      const float xr = lane_bcast(xv, bperm[r]);
      float g0 = __builtin_fmaf(xr, wih[0], acc[0][r]) + bia[0];   // i (col lc)
      float g1 = __builtin_fmaf(xr, wih[1], acc[1][r]) + bia[1];   // i (col lc+16)
      float g2 = __builtin_fmaf(xr, wih[2], acc[2][r]) + bia[2];   // f
      float g3 = __builtin_fmaf(xr, wih[3], acc[3][r]) + bia[3];   // f
      float g4 = __builtin_fmaf(xr, wih[4], acc[4][r]) + bia[4];   // g
      float g5 = __builtin_fmaf(xr, wih[5], acc[5][r]) + bia[5];   // g
      float g6 = __builtin_fmaf(xr, wih[6], acc[6][r]) + bia[6];   // o
      float g7 = __builtin_fmaf(xr, wih[7], acc[7][r]) + bia[7];   // o
      cs0[r] = sigm(g2) * cs0[r] + sigm(g0) * ftanh(g4);
      cs1[r] = sigm(g3) * cs1[r] + sigm(g1) * ftanh(g5);
      const float h0 = sigm(g6) * ftanh(cs0[r]);
      const float h1 = sigm(g7) * ftanh(cs1[r]);
      const int M = r + kb;
      hbuf[M * 32 + lc]      = (_Float16)h0;
      hbuf[M * 32 + lc + 16] = (_Float16)h1;
    }
    __syncthreads();
    xv = xnext;
  }

  // ===== emit forward hidden at last step: Xm[b][s=j][c] ===================
#pragma unroll
  for (int r = 0; r < 8; ++r) {
    const int M = r + kb;
    const size_t o = (size_t)(rowbase + M) * (SEQM * C) + c;
    Xm[o + (size_t)lc * C]        = hbuf[M * 32 + lc];
    Xm[o + (size_t)(lc + 16) * C] = hbuf[M * 32 + lc + 16];
  }

  // ===== backward direction: single zero-state cell on x[:, T-1] ===========
  // (after the loop, xv == x[:, T-1])
#pragma unroll
  for (int r = 0; r < 8; ++r) {
    const float xr = lane_bcast(xv, bperm[r]);
    float g0 = __builtin_fmaf(xr, wihb[0], biab[0]);
    float g1 = __builtin_fmaf(xr, wihb[1], biab[1]);
    float g4 = __builtin_fmaf(xr, wihb[4], biab[4]);
    float g5 = __builtin_fmaf(xr, wihb[5], biab[5]);
    float g6 = __builtin_fmaf(xr, wihb[6], biab[6]);
    float g7 = __builtin_fmaf(xr, wihb[7], biab[7]);
    const float c0 = sigm(g0) * ftanh(g4);           // f-gate * 0 drops out
    const float c1 = sigm(g1) * ftanh(g5);
    const float h0 = sigm(g6) * ftanh(c0);
    const float h1 = sigm(g7) * ftanh(c1);
    const int M = r + kb;
    const size_t o = (size_t)(rowbase + M) * (SEQM * C) + c;
    Xm[o + (size_t)(32 + lc) * C] = (_Float16)h0;
    Xm[o + (size_t)(48 + lc) * C] = (_Float16)h1;
  }
}

// ===========================================================================
// Kernel 2: merge BiLSTM over [B][64][C] f16 input
// grid = B/16, block = 32 (one wave per 16 batch rows)
// ===========================================================================
__global__ __launch_bounds__(32) void k_merge_lstm(
    const _Float16* __restrict__ Xm,    // [B,64,C] f16
    const float* __restrict__ WihMF,    // [128,32]
    const float* __restrict__ WhhMF,    // [128,32]
    const float* __restrict__ bihMF,    // [128]
    const float* __restrict__ bhhMF,    // [128]
    const float* __restrict__ WihMB,    // [128,32]
    const float* __restrict__ bihMB,    // [128]
    const float* __restrict__ bhhMB,    // [128]
    float* __restrict__ Mg)             // [B,64] merged hidden (f32)
{
  const int rowbase = blockIdx.x * 16;
  const int lane    = threadIdx.x & 31;
  const int lc      = lane & 15;
  const int kb      = (lane >> 4) * 8;

  __shared__ _Float16 hbuf[16 * 32];

  v16h bh[8], bx[8];
#pragma unroll
  for (int n = 0; n < 8; ++n) {
    bh[n] = load_btile(WhhMF, n, lc, kb);
    bx[n] = load_btile(WihMF, n, lc, kb);
  }
  float bia[8];
#pragma unroll
  for (int n = 0; n < 8; ++n) {
    const int col = n * 16 + lc;
    bia[n] = bihMF[col] + bhhMF[col];
  }

  float cs0[8], cs1[8];
#pragma unroll
  for (int r = 0; r < 8; ++r) {
    cs0[r] = 0.0f; cs1[r] = 0.0f;
    const int M = r + kb;
    hbuf[M * 32 + lc]      = (_Float16)0.0f;
    hbuf[M * 32 + lc + 16] = (_Float16)0.0f;
  }
  __syncthreads();

  const size_t xrow = (size_t)(rowbase + lc) * (SEQM * C);

  // A_x loader: 16x32 f16 tile at merge step s (two global_load_b128)
  auto load_ax = [&](int s) -> v16h {
    const _Float16* xp = Xm + xrow + (size_t)s * C + kb;
    v8h x0 = *(const v8h*)(xp);
    v8h x1 = *(const v8h*)(xp + 16);
    v16h ax;
#pragma unroll
    for (int q = 0; q < 8; ++q) { ax[q] = x0[q]; ax[8 + q] = x1[q]; }
    return ax;
  };

  // ===== forward scan over 64 merge steps (A_x pipelined) ==================
  v16h ax = load_ax(0);
  for (int s = 0; s < SEQM; ++s) {
    const int sn = (s + 1 < SEQM) ? (s + 1) : (SEQM - 1);
    v16h axn = load_ax(sn);

    // A_h from LDS
    const _Float16* hrow = hbuf + lc * 32 + kb;
    v8h a0 = *(const v8h*)(hrow);
    v8h a1 = *(const v8h*)(hrow + 16);
    v16h ah;
#pragma unroll
    for (int q = 0; q < 8; ++q) { ah[q] = a0[q]; ah[8 + q] = a1[q]; }

    // gates = x_s @ Wih^T + h @ Whh^T : chained WMMAs per N-tile
    v8f acc[8];
#pragma unroll
    for (int n = 0; n < 8; ++n) {
      v8f z = {};
      v8f p = __builtin_amdgcn_wmma_f32_16x16x32_f16(
          false, ax, false, bx[n], (short)0, z, false, false);
      acc[n] = __builtin_amdgcn_wmma_f32_16x16x32_f16(
          false, ah, false, bh[n], (short)0, p, false, false);
    }
    __syncthreads();

#pragma unroll
    for (int r = 0; r < 8; ++r) {
      float g0 = acc[0][r] + bia[0];
      float g1 = acc[1][r] + bia[1];
      float g2 = acc[2][r] + bia[2];
      float g3 = acc[3][r] + bia[3];
      float g4 = acc[4][r] + bia[4];
      float g5 = acc[5][r] + bia[5];
      float g6 = acc[6][r] + bia[6];
      float g7 = acc[7][r] + bia[7];
      cs0[r] = sigm(g2) * cs0[r] + sigm(g0) * ftanh(g4);
      cs1[r] = sigm(g3) * cs1[r] + sigm(g1) * ftanh(g5);
      const float h0 = sigm(g6) * ftanh(cs0[r]);
      const float h1 = sigm(g7) * ftanh(cs1[r]);
      const int M = r + kb;
      hbuf[M * 32 + lc]      = (_Float16)h0;
      hbuf[M * 32 + lc + 16] = (_Float16)h1;
      if (s == SEQM - 1) {                       // emit forward half
        Mg[(size_t)(rowbase + M) * 64 + lc]      = h0;
        Mg[(size_t)(rowbase + M) * 64 + lc + 16] = h1;
      }
    }
    __syncthreads();
    ax = axn;                                    // after loop: ax == step 63
  }

  // ===== backward direction: one zero-state cell on step 63 ================
  v16h bb[8];
#pragma unroll
  for (int n = 0; n < 8; ++n) bb[n] = load_btile(WihMB, n, lc, kb);
  float biab[8];
#pragma unroll
  for (int n = 0; n < 8; ++n) {
    const int col = n * 16 + lc;
    biab[n] = bihMB[col] + bhhMB[col];
  }

  v8f acc[8];
#pragma unroll
  for (int n = 0; n < 8; ++n) {
    v8f z = {};
    acc[n] = __builtin_amdgcn_wmma_f32_16x16x32_f16(
        false, ax, false, bb[n], (short)0, z, false, false);
  }

#pragma unroll
  for (int r = 0; r < 8; ++r) {
    float g0 = acc[0][r] + biab[0];
    float g1 = acc[1][r] + biab[1];
    float g4 = acc[4][r] + biab[4];
    float g5 = acc[5][r] + biab[5];
    float g6 = acc[6][r] + biab[6];
    float g7 = acc[7][r] + biab[7];
    const float c0 = sigm(g0) * ftanh(g4);
    const float c1 = sigm(g1) * ftanh(g5);
    const int M = r + kb;
    Mg[(size_t)(rowbase + M) * 64 + 32 + lc] = sigm(g6) * ftanh(c0);
    Mg[(size_t)(rowbase + M) * 64 + 48 + lc] = sigm(g7) * ftanh(c1);
  }
}

// ===========================================================================
// Kernel 3: out = sigmoid(Mg @ fc_w^T + fc_b)   [128 x 64] x [64 x 25]
// ===========================================================================
__global__ void k_fc(const float* __restrict__ Mg,
                     const float* __restrict__ W,    // [25,64]
                     const float* __restrict__ bv,   // [25]
                     float* __restrict__ out)        // [128,25]
{
  const int idx = blockIdx.x * blockDim.x + threadIdx.x;
  if (idx >= B * NCLS) return;
  const int b = idx / NCLS;
  const int k = idx % NCLS;
  const float* m = Mg + (size_t)b * 64;
  const float* w = W + (size_t)k * 64;
  float acc = bv[k];
#pragma unroll
  for (int j = 0; j < 64; ++j) acc = __builtin_fmaf(m[j], w[j], acc);
  out[idx] = sigm(acc);
}

// ===========================================================================
extern "C" void kernel_launch(void* const* d_in, const int* in_sizes, int n_in,
                              void* d_out, int out_size, void* d_ws, size_t ws_size,
                              hipStream_t stream) {
  (void)in_sizes; (void)n_in; (void)out_size; (void)ws_size;

  const float* x       = (const float*)d_in[0];
  const float* Wih_cf  = (const float*)d_in[1];
  const float* Whh_cf  = (const float*)d_in[2];
  const float* bih_cf  = (const float*)d_in[3];
  const float* bhh_cf  = (const float*)d_in[4];
  const float* Wih_cb  = (const float*)d_in[5];
  /* Whh_cb (d_in[6]) unused: backward starts from zero state */
  const float* bih_cb  = (const float*)d_in[7];
  const float* bhh_cb  = (const float*)d_in[8];
  const float* Wih_mf  = (const float*)d_in[9];
  const float* Whh_mf  = (const float*)d_in[10];
  const float* bih_mf  = (const float*)d_in[11];
  const float* bhh_mf  = (const float*)d_in[12];
  const float* Wih_mb  = (const float*)d_in[13];
  /* Whh_mb (d_in[14]) unused: backward starts from zero state */
  const float* bih_mb  = (const float*)d_in[15];
  const float* bhh_mb  = (const float*)d_in[16];
  const float* fc_w    = (const float*)d_in[17];
  const float* fc_b    = (const float*)d_in[18];

  // workspace layout
  _Float16* Xm = (_Float16*)d_ws;                                   // 512 KB
  float*    Mg = (float*)((char*)d_ws + (size_t)B * SEQM * C * 2);  //  32 KB

  // 1) 256 independent wave-chains: per-channel BiLSTM
  k_chan_lstm<<<dim3(C, B / 16), 32, 0, stream>>>(
      x, Wih_cf, Whh_cf, bih_cf, bhh_cf, Wih_cb, bih_cb, bhh_cb, Xm);

  // 2) merge BiLSTM over [B,64,C]
  k_merge_lstm<<<dim3(B / 16), 32, 0, stream>>>(
      Xm, Wih_mf, Whh_mf, bih_mf, bhh_mf, Wih_mb, bih_mb, bhh_mb, Mg);

  // 3) FC + sigmoid
  k_fc<<<dim3((B * NCLS + 255) / 256), 256, 0, stream>>>(Mg, fc_w, fc_b,
                                                         (float*)d_out);
}
